// BiauralProcessor_30580167147906
// MI455X (gfx1250) — matrix operand order repaired and checked by reading, not compiled, not run on hardware
//
#include <hip/hip_runtime.h>

// MI455X / gfx1250 binaural ITD+ILD kernel.
// ITD via banded cross-correlation tiles computed with v_wmma_f32_16x16x32_f16,
// ILD fused from the same global loads (exact f32 row sums + cross-lane shfl).

typedef __attribute__((ext_vector_type(16))) _Float16 v16h;
typedef __attribute__((ext_vector_type(8)))  float    v8f;
typedef __attribute__((ext_vector_type(4)))  float    vf4;

#define B_    16
#define T_    16000
#define F_    64
#define NITD  16
#define NILD  8
#define TPB   (T_ / 16)          // time tiles per batch = 1000
#define WPB   8                  // waves per block (256 threads, wave32)

// Load 8 floats at p0 -> dst[0..7], 8 floats at p1 -> dst[8..15] (f32->f16).
// Returns the f32 sum of all 16 loaded values (used for the ILD row sums).
__device__ __forceinline__ float pack16(v16h& dst, const float* p0, const float* p1) {
    vf4 a = *(const vf4*)(p0);
    vf4 b = *(const vf4*)(p0 + 4);
    vf4 c = *(const vf4*)(p1);
    vf4 d = *(const vf4*)(p1 + 4);
    dst[0]  = (_Float16)a.x; dst[1]  = (_Float16)a.y; dst[2]  = (_Float16)a.z; dst[3]  = (_Float16)a.w;
    dst[4]  = (_Float16)b.x; dst[5]  = (_Float16)b.y; dst[6]  = (_Float16)b.z; dst[7]  = (_Float16)b.w;
    dst[8]  = (_Float16)c.x; dst[9]  = (_Float16)c.y; dst[10] = (_Float16)c.z; dst[11] = (_Float16)c.w;
    dst[12] = (_Float16)d.x; dst[13] = (_Float16)d.y; dst[14] = (_Float16)d.z; dst[15] = (_Float16)d.w;
    return (a.x + a.y + a.z + a.w) + (b.x + b.y + b.z + b.w) +
           (c.x + c.y + c.z + c.w) + (d.x + d.y + d.z + d.w);
}

__global__ __launch_bounds__(256) void
biaural_itd_ild_kernel(const float* __restrict__ Lsp,
                       const float* __restrict__ Rsp,
                       float* __restrict__ out_itd,   // [B,T,16]
                       float* __restrict__ out_ild) { // [B,T,8]
    __shared__ float ldsG[WPB][256];   // per-wave 16x16 f32 tile of G = R * L^T
    __shared__ float ldsH[WPB][256];   // per-wave 16x16 f32 tile of H = L * R^T

    const int wave = threadIdx.x >> 5;
    const int lane = threadIdx.x & 31;
    const int hi   = lane >> 4;        // 0: lanes 0-15, 1: lanes 16-31
    const int m    = lane & 15;

    const int tile = blockIdx.x * WPB + wave;   // 0 .. 15999
    const int b    = tile / TPB;
    const int t0   = (tile % TPB) * 16;

    const float* Lb = Lsp + (size_t)b * (T_ * F_);
    const float* Rb = Rsp + (size_t)b * (T_ * F_);

    // ---- A operands: rows t0..t0+15 of R (for G) and L (for H), f16 A-layout.
    // Lane layout (16-bit A 16x32): lane<16 holds K {kk*32+0..7, kk*32+16..23},
    // lane>=16 holds K {kk*32+8..15, kk*32+24..31} of row M = lane&15.
    v16h aL[2], aR[2];
    float sL = 0.f, sR = 0.f;
    {
        const float* lrow = Lb + (size_t)(t0 + m) * F_;
        const float* rrow = Rb + (size_t)(t0 + m) * F_;
#pragma unroll
        for (int kk = 0; kk < 2; ++kk) {
            const int f1 = kk * 32 + hi * 8;
            sL += pack16(aL[kk], lrow + f1, lrow + f1 + 16);
            sR += pack16(aR[kk], rrow + f1, rrow + f1 + 16);
        }
    }

    // ---- ILD (exact f32): lanes i and i+16 hold disjoint halves of row i's 64 ch.
    {
        const float fl = sL + __shfl_xor(sL, 16, 32);
        const float fr = sR + __shfl_xor(sR, 16, 32);
        if (lane < 16) {
            const float ild = (fl - fr) / (fl + fr + 1e-6f);
            float o[NILD];
#pragma unroll
            for (int j = 0; j < NILD; ++j) {
                const float pref = -1.f + (2.f / 7.f) * (float)j;
                const float z = (ild - pref) * (1.f / 0.3f);
                o[j] = __expf(-0.5f * z * z);
            }
            float* p = out_ild + ((size_t)b * T_ + (t0 + lane)) * NILD;
            vf4 v0 = {o[0], o[1], o[2], o[3]};
            vf4 v1 = {o[4], o[5], o[6], o[7]};
            *(vf4*)(p)     = v0;
            *(vf4*)(p + 4) = v1;
        }
    }

    // ITD_DELAYS = round(linspace(-32,32,16)):
    //  j=0..7  : -32,-28,-23,-19,-15,-11,-6,-2  -> diagonals of H
    //  j=8..15 :   2,  6, 11, 15, 19, 23, 28,32 -> diagonals of G
    const int NEGMAG[8] = {32, 28, 23, 19, 15, 11, 6, 2};
    const int POSMAG[8] = {2, 6, 11, 15, 19, 23, 28, 32};

    float outA[8];
#pragma unroll
    for (int j = 0; j < 8; ++j) outA[j] = 0.f;  // zero = padding region t < |d|

    // ---- 3 column tiles c0 = t0-32, t0-16, t0 cover every needed diagonal once.
#pragma unroll
    for (int c = 0; c < 3; ++c) {
        const int c0 = t0 - 32 + 16 * c;
        int brow = c0 + m;
        if (brow < 0) brow = 0;  // clamped rows are never consumed (gather masks them)

        // B operands, f16 B-layout (32x16): lane = column n, same K order as A
        // (lanes 0-15: K = kk*32+{0..7,16..23}, lanes 16-31: K = kk*32+{8..15,24..31}).
        v16h bL[2], bR[2];
        {
            const float* blp = Lb + (size_t)brow * F_;
            const float* brp = Rb + (size_t)brow * F_;
#pragma unroll
            for (int kk = 0; kk < 2; ++kk) {
                const int f0 = kk * 32 + hi * 8;
                (void)pack16(bL[kk], blp + f0, blp + f0 + 16);
                (void)pack16(bR[kk], brp + f0, brp + f0 + 16);
            }
        }

        // G[t, c0+n] = sum_f R[t,f] L[c0+n,f] ; H[t, c0+n] = sum_f L[t,f] R[c0+n,f]
        v8f accG = {};
        accG = __builtin_amdgcn_wmma_f32_16x16x32_f16(false, aR[0], false, bL[0],
                                                      (short)0, accG, false, false);
        accG = __builtin_amdgcn_wmma_f32_16x16x32_f16(false, aR[1], false, bL[1],
                                                      (short)0, accG, false, false);
        v8f accH = {};
        accH = __builtin_amdgcn_wmma_f32_16x16x32_f16(false, aL[0], false, bR[0],
                                                      (short)0, accH, false, false);
        accH = __builtin_amdgcn_wmma_f32_16x16x32_f16(false, aL[1], false, bR[1],
                                                      (short)0, accH, false, false);

        // C/D layout: VGPR r, lane<16 -> row r, lane>=16 -> row 8+r, col = lane&15.
#pragma unroll
        for (int r = 0; r < 8; ++r) {
            const int mm = hi * 8 + r;
            ldsG[wave][mm * 16 + m] = accG[r];
            ldsH[wave][mm * 16 + m] = accH[r];
        }
        __syncthreads();

        // Gather diagonals: out[t0+i, d] sits at tile entry (row i, col i-d+32-16c).
        if (lane < 16) {
            const int i = lane;
#pragma unroll
            for (int j = 0; j < 8; ++j) {
                const int d = NEGMAG[j];
                const int n = i - d + 32 - 16 * c;
                if (n >= 0 && n < 16 && (t0 + i - d) >= 0)
                    outA[j] = ldsH[wave][i * 16 + n];
            }
        } else {
            const int i = lane - 16;
#pragma unroll
            for (int j = 0; j < 8; ++j) {
                const int d = POSMAG[j];
                const int n = i - d + 32 - 16 * c;
                if (n >= 0 && n < 16 && (t0 + i - d) >= 0)
                    outA[j] = ldsG[wave][i * 16 + n];
            }
        }
        __syncthreads();  // protect LDS reuse next iteration
    }

    // ---- Store ITD: lane<16 -> j 0..7 of row (t0+lane); lane>=16 -> j 8..15.
    {
        const int i = lane & 15;
        float* p = out_itd + ((size_t)b * T_ + (t0 + i)) * NITD + hi * 8;
        vf4 v0 = {outA[0], outA[1], outA[2], outA[3]};
        vf4 v1 = {outA[4], outA[5], outA[6], outA[7]};
        *(vf4*)(p)     = v0;
        *(vf4*)(p + 4) = v1;
    }
}

extern "C" void kernel_launch(void* const* d_in, const int* in_sizes, int n_in,
                              void* d_out, int out_size, void* d_ws, size_t ws_size,
                              hipStream_t stream) {
    (void)in_sizes; (void)n_in; (void)out_size; (void)d_ws; (void)ws_size;
    const float* left  = (const float*)d_in[0];
    const float* right = (const float*)d_in[1];
    float* itd = (float*)d_out;                                // [16,16000,16]
    float* ild = itd + (size_t)B_ * T_ * NITD;                 // [16,16000,8]

    const int total_waves = B_ * TPB;          // 16000 tiles, 1 wave each
    const int blocks = total_waves / WPB;      // 2000 blocks x 256 threads
    hipLaunchKernelGGL(biaural_itd_ild_kernel, dim3(blocks), dim3(256), 0, stream,
                       left, right, itd, ild);
}
